// NDNRefinement_81449759801400
// MI455X (gfx1250) — compile-verified, matrix operand order checked by reading, not run on hardware
//
#include <hip/hip_runtime.h>
#include <hip/hip_bf16.h>

typedef __bf16 bf16_t;
typedef __attribute__((ext_vector_type(16))) __bf16 v16bf;
typedef __attribute__((ext_vector_type(8)))  float   v8f;

union Frag {
  uint4 u[2];   // 32 bytes = 16 bf16
  v16bf v;
};

// ---------------------------------------------------------------------------
// Generic WMMA GEMM:  C[M,N] = lrelu(A[M,K] @ Bt[N,K]^T + bias)
//   AM==0: A is a dense bf16 row-major matrix (row stride K)
//   AM==1: A row m is the concat [obj[s_idx[m]], pred[m], obj[o_idx[m]]], K=3*din.
//          Since din % 32 == 0, each 32-wide K-slab lies in exactly one segment,
//          so the K loop is split into 3 branch-free segment loops.
//   EM==0: store bf16 to outb (ld = ldo)
//   EM==1: scatter epilogue: n<H -> atomicAdd pooled[s_idx[m]*H+n]
//                            H<=n<H+Dout -> predout[m*Dout + n-H] (bf16)
//                            else -> atomicAdd pooled[o_idx[m]*H + n-H-Dout]
//   EM==2: store fp32 to outf (ld = ldo)
// Bt is padded (zero-filled) to a multiple of 128 rows, so B-fragment loads
// need no guards: inner loop is exactly 16 b128 loads + 16 WMMAs.
// Block: 256 thr = 8 waves (4 M x 2 N); block tile 256x128; wave tile 64x64.
// ---------------------------------------------------------------------------
template<int AM, int EM>
__global__ __launch_bounds__(256)
void gemm_wmma(const bf16_t* __restrict__ A,
               const bf16_t* __restrict__ objv,
               const bf16_t* __restrict__ predv,
               const int* __restrict__ sidx,
               const int* __restrict__ oidx,
               int din,
               const bf16_t* __restrict__ Bt,
               const float* __restrict__ bias,
               int M, int N, int K,
               bf16_t* __restrict__ outb, int ldo,
               float* __restrict__ pooled,
               bf16_t* __restrict__ predout,
               float* __restrict__ outf,
               int H, int Dout)
{
  const int lane = threadIdx.x & 31;
  const int wave = threadIdx.x >> 5;
  const int wm   = wave & 3;          // 0..3  (M direction)
  const int wn   = wave >> 2;         // 0..1  (N direction)
  const int bM   = blockIdx.x * 256;
  const int bN   = blockIdx.y * 128;
  const int lrow = lane & 15;         // row within a 16-tile
  const int kb   = (lane >> 4) << 3;  // 0 or 8: K-chunk base within 16-wide half

  v8f acc[4][4];
#pragma unroll
  for (int i = 0; i < 4; ++i)
#pragma unroll
    for (int j = 0; j < 4; ++j)
#pragma unroll
      for (int v = 0; v < 8; ++v) acc[i][j][v] = 0.0f;

  // Per-M-tile row pointers (clamped for ragged M)
  const bf16_t* aPtr[4];
  const bf16_t* sPtr[4];
  const bf16_t* pPtr[4];
  const bf16_t* oPtr[4];
#pragma unroll
  for (int mt = 0; mt < 4; ++mt) {
    int m = bM + wm * 64 + mt * 16 + lrow;
    if (m > M - 1) m = M - 1;
    if (AM == 0) {
      aPtr[mt] = A + (size_t)m * K;
    } else {
      sPtr[mt] = objv  + (size_t)sidx[m] * din;
      pPtr[mt] = predv + (size_t)m       * din;
      oPtr[mt] = objv  + (size_t)oidx[m] * din;
    }
  }
  // Per-N-tile Bt row pointers (Bt rows padded to multiple of 128: no guard)
  const bf16_t* bPtr[4];
#pragma unroll
  for (int nt = 0; nt < 4; ++nt) {
    int n = bN + wn * 64 + nt * 16 + lrow;
    bPtr[nt] = Bt + (size_t)n * K;
  }

  constexpr int NSEG = (AM == 1) ? 3 : 1;
#pragma unroll
  for (int seg = 0; seg < NSEG; ++seg) {
    const int segLen = (AM == 1) ? din : K;
    const int kbase  = seg * segLen;
    const bf16_t* ra[4];
#pragma unroll
    for (int mt = 0; mt < 4; ++mt) {
      if (AM == 0) ra[mt] = aPtr[mt];
      else         ra[mt] = (seg == 0) ? sPtr[mt] : ((seg == 1) ? pPtr[mt] : oPtr[mt]);
    }
    for (int kk = 0; kk < segLen; kk += 32) {
      Frag fa[4], fb[4];
#pragma unroll
      for (int mt = 0; mt < 4; ++mt) {
        const uint4* r = reinterpret_cast<const uint4*>(ra[mt] + kk + kb);
        fa[mt].u[0] = r[0];       // K = kk+kb .. +7
        fa[mt].u[1] = r[2];       // K = kk+kb+16 .. +23
      }
#pragma unroll
      for (int nt = 0; nt < 4; ++nt) {
        const uint4* r = reinterpret_cast<const uint4*>(bPtr[nt] + kbase + kk + kb);
        fb[nt].u[0] = r[0];
        fb[nt].u[1] = r[2];
      }
#pragma unroll
      for (int mt = 0; mt < 4; ++mt)
#pragma unroll
        for (int nt = 0; nt < 4; ++nt)
          acc[mt][nt] = __builtin_amdgcn_wmma_f32_16x16x32_bf16(
              false, fa[mt].v, false, fb[nt].v, (short)0, acc[mt][nt], false, false);
    }
  }

  // Epilogue: bias + leaky-relu + store/scatter
#pragma unroll
  for (int mt = 0; mt < 4; ++mt) {
#pragma unroll
    for (int nt = 0; nt < 4; ++nt) {
      const int n = bN + wn * 64 + nt * 16 + lrow;
      const float bia = (n < N) ? bias[n] : 0.0f;
      const int mbase = bM + wm * 64 + mt * 16 + ((lane >> 4) << 3);
#pragma unroll
      for (int v = 0; v < 8; ++v) {
        const int m = mbase + v;
        float x = acc[mt][nt][v] + bia;
        x = (x > 0.0f) ? x : 0.2f * x;
        if (EM == 0) {
          if (m < M && n < N) outb[(size_t)m * ldo + n] = (bf16_t)x;
        } else if (EM == 2) {
          if (m < M && n < N) outf[(size_t)m * ldo + n] = x;
        } else {
          if (m < M && n < N) {
            if (n < H) {
              unsafeAtomicAdd(pooled + (size_t)sidx[m] * H + n, x);
            } else if (n < H + Dout) {
              predout[(size_t)m * Dout + (n - H)] = (bf16_t)x;
            } else {
              unsafeAtomicAdd(pooled + (size_t)oidx[m] * H + (n - H - Dout), x);
            }
          }
        }
      }
    }
  }
}

// ---------------------------------------------------------------------------
// Small helper kernels
// ---------------------------------------------------------------------------
__global__ void k_cvt(const float* __restrict__ in, bf16_t* __restrict__ out, int n) {
  int i = blockIdx.x * blockDim.x + threadIdx.x;
  if (i < n) out[i] = (bf16_t)in[i];
}

// Bt[n*Kpad + k] = (k<din && n<dout) ? W[k*dout + n] : 0   (bf16)
// Nrows is padded to a multiple of 128 and the padding rows are zero-filled.
__global__ void k_transpose(const float* __restrict__ W, bf16_t* __restrict__ Bt,
                            int din, int dout, int Kpad, int Nrows) {
  int k = blockIdx.x * blockDim.x + threadIdx.x;
  int n = blockIdx.y;
  if (k >= Kpad || n >= Nrows) return;
  float v = (k < din && n < dout) ? W[(size_t)k * dout + n] : 0.0f;
  Bt[(size_t)n * Kpad + k] = (bf16_t)v;
}

__global__ void k_count(const int* __restrict__ s, const int* __restrict__ o,
                        float* __restrict__ counts, int T) {
  int t = blockIdx.x * blockDim.x + threadIdx.x;
  if (t < T) {
    unsafeAtomicAdd(counts + s[t], 1.0f);
    unsafeAtomicAdd(counts + o[t], 1.0f);
  }
}

__global__ void k_inv(const float* __restrict__ counts, float* __restrict__ inv, int O) {
  int i = blockIdx.x * blockDim.x + threadIdx.x;
  if (i < O) inv[i] = 1.0f / fmaxf(counts[i], 1.0f);
}

__global__ void k_pooldiv(const float* __restrict__ pooled, const float* __restrict__ inv,
                          bf16_t* __restrict__ out, int O, int H) {
  int row = blockIdx.x;
  int col = blockIdx.y * 128 + threadIdx.x;
  if (col < H) out[(size_t)row * H + col] = (bf16_t)(pooled[(size_t)row * H + col] * inv[row]);
}

// xenc row: [boxes(4) | obj(128) | zeros(pad to 160)]
__global__ void k_concat(const float* __restrict__ boxes, const bf16_t* __restrict__ obj,
                         bf16_t* __restrict__ xenc, int O) {
  int row = blockIdx.x;
  int col = threadIdx.x;
  if (col >= 160) return;
  float v;
  if (col < 4)        v = boxes[(size_t)row * 4 + col];
  else if (col < 132) v = (float)obj[(size_t)row * 128 + (col - 4)];
  else                v = 0.0f;
  xenc[(size_t)row * 160 + col] = (bf16_t)v;
}

// ---------------------------------------------------------------------------
extern "C" void kernel_launch(void* const* d_in, const int* in_sizes, int n_in,
                              void* d_out, int out_size, void* d_ws, size_t ws_size,
                              hipStream_t stream) {
  (void)n_in; (void)out_size; (void)ws_size;

  const float* obj_vecs   = (const float*)d_in[0];
  const float* pred_vecs  = (const float*)d_in[1];
  const float* pred_boxes = (const float*)d_in[2];
  const int*   s_idx      = (const int*)d_in[3];
  const int*   o_idx      = (const int*)d_in[4];

  const int O = in_sizes[0] / 64;   // obj_vecs is O x 64
  const int T = in_sizes[3];        // s_idx length

  // params: 4 layers x {w1a,b1a,w1b,b1b,w2a,b2a,w2b,b2b} then 3 enc (w,b)
  const float* LW[4][8];
  int q = 5;
  for (int l = 0; l < 4; ++l)
    for (int j = 0; j < 8; ++j) LW[l][j] = (const float*)d_in[q++];
  const float* encw[3]; const float* encb[3];
  for (int e = 0; e < 3; ++e) { encw[e] = (const float*)d_in[q++]; encb[e] = (const float*)d_in[q++]; }

  // ---- workspace carving (256B aligned) ----
  char* base = (char*)d_ws;
  size_t off = 0;
  auto carve = [&](size_t bytes) -> void* {
    void* r = base + off;
    off += (bytes + 255) & ~(size_t)255;
    return r;
  };
  bf16_t* objA    = (bf16_t*)carve((size_t)O * 128 * 2);
  bf16_t* objB    = (bf16_t*)carve((size_t)O * 128 * 2);
  bf16_t* predA   = (bf16_t*)carve((size_t)T * 128 * 2);
  bf16_t* predB   = (bf16_t*)carve((size_t)T * 128 * 2);
  bf16_t* h1      = (bf16_t*)carve((size_t)T * 512 * 2);
  float*  pooled  = (float*) carve((size_t)O * 512 * 4);
  bf16_t* pooledb = (bf16_t*)carve((size_t)O * 512 * 2);
  bf16_t* t1      = (bf16_t*)carve((size_t)O * 512 * 2);
  float*  counts  = (float*) carve((size_t)O * 4);
  float*  invc    = (float*) carve((size_t)O * 4);
  bf16_t* Btw     = (bf16_t*)carve((size_t)1152 * 512 * 2);
  bf16_t* xenc    = (bf16_t*)carve((size_t)O * 160 * 2);

  // Transpose weights to (Nrows x Kpad) bf16, Nrows padded to multiple of 128
  auto transW = [&](const float* W, int din_w, int dout_w, int Kpad) {
    int Nrows = ((dout_w + 127) / 128) * 128;
    dim3 g((Kpad + 255) / 256, Nrows);
    k_transpose<<<g, 256, 0, stream>>>(W, Btw, din_w, dout_w, Kpad, Nrows);
  };

  const int DIN[4] = {64, 128, 128, 128};
  const int HH[4]  = {512, 512, 512, 128};
  const int DO_[4] = {128, 128, 128, 128};

  // Initial bf16 conversions + counts (same for all layers)
  k_cvt<<<(O * 64 + 255) / 256, 256, 0, stream>>>(obj_vecs, objA, O * 64);
  k_cvt<<<(T * 64 + 255) / 256, 256, 0, stream>>>(pred_vecs, predA, T * 64);
  hipMemsetAsync(counts, 0, (size_t)O * 4, stream);
  k_count<<<(T + 255) / 256, 256, 0, stream>>>(s_idx, o_idx, counts, T);
  k_inv<<<(O + 255) / 256, 256, 0, stream>>>(counts, invc, O);

  bf16_t* objCur = objA;  bf16_t* objNxt = objB;
  bf16_t* predCur = predA; bf16_t* predNxt = predB;

  for (int l = 0; l < 4; ++l) {
    const int din = DIN[l], H = HH[l], Do = DO_[l];

    // mm1: h1 = lrelu(concat @ w1a + b1a)      (T x H), gathered A
    transW(LW[l][0], 3 * din, H, 3 * din);
    {
      dim3 g((T + 255) / 256, (H + 127) / 128);
      gemm_wmma<1, 0><<<g, 256, 0, stream>>>(
          (const bf16_t*)nullptr, objCur, predCur, s_idx, o_idx, din,
          Btw, LW[l][1], T, H, 3 * din,
          h1, H, (float*)nullptr, (bf16_t*)nullptr, (float*)nullptr, 0, 0);
    }

    // mm2 + scatter: h2 = lrelu(h1 @ w1b + b1b); split into pooled/pred
    hipMemsetAsync(pooled, 0, (size_t)O * H * 4, stream);
    transW(LW[l][2], H, 2 * H + Do, H);
    {
      const int N2 = 2 * H + Do;
      dim3 g((T + 255) / 256, (N2 + 127) / 128);
      gemm_wmma<0, 1><<<g, 256, 0, stream>>>(
          h1, (const bf16_t*)nullptr, (const bf16_t*)nullptr, s_idx, o_idx, 0,
          Btw, LW[l][3], T, N2, H,
          (bf16_t*)nullptr, 0, pooled, predNxt, (float*)nullptr, H, Do);
    }

    // pooled / max(counts,1) -> bf16
    {
      dim3 g(O, H / 128);
      k_pooldiv<<<g, 128, 0, stream>>>(pooled, invc, pooledb, O, H);
    }

    // mm3: t1 = lrelu(pooledb @ w2a + b2a)    (O x H)
    transW(LW[l][4], H, H, H);
    {
      dim3 g((O + 255) / 256, (H + 127) / 128);
      gemm_wmma<0, 0><<<g, 256, 0, stream>>>(
          pooledb, (const bf16_t*)nullptr, (const bf16_t*)nullptr, s_idx, o_idx, 0,
          Btw, LW[l][5], O, H, H,
          t1, H, (float*)nullptr, (bf16_t*)nullptr, (float*)nullptr, 0, 0);
    }

    // mm4: objNxt = lrelu(t1 @ w2b + b2b)     (O x Do)
    transW(LW[l][6], H, Do, H);
    {
      dim3 g((O + 255) / 256, (Do + 127) / 128);
      gemm_wmma<0, 0><<<g, 256, 0, stream>>>(
          t1, (const bf16_t*)nullptr, (const bf16_t*)nullptr, s_idx, o_idx, 0,
          Btw, LW[l][7], O, Do, H,
          objNxt, Do, (float*)nullptr, (bf16_t*)nullptr, (float*)nullptr, 0, 0);
    }

    { bf16_t* t = objCur; objCur = objNxt; objNxt = t; }
    { bf16_t* t = predCur; predCur = predNxt; predNxt = t; }
  }

  // enc head: x = concat(boxes, obj) (O x 132 -> padded 160)
  k_concat<<<O, 192, 0, stream>>>(pred_boxes, objCur, xenc, O);

  transW(encw[0], 132, 128, 160);
  {
    dim3 g((O + 255) / 256, 1);
    gemm_wmma<0, 0><<<g, 256, 0, stream>>>(
        xenc, (const bf16_t*)nullptr, (const bf16_t*)nullptr, s_idx, o_idx, 0,
        Btw, encb[0], O, 128, 160,
        t1, 128, (float*)nullptr, (bf16_t*)nullptr, (float*)nullptr, 0, 0);
  }
  transW(encw[1], 128, 128, 128);
  {
    dim3 g((O + 255) / 256, 1);
    gemm_wmma<0, 0><<<g, 256, 0, stream>>>(
        t1, (const bf16_t*)nullptr, (const bf16_t*)nullptr, s_idx, o_idx, 0,
        Btw, encb[1], O, 128, 128,
        pooledb, 128, (float*)nullptr, (bf16_t*)nullptr, (float*)nullptr, 0, 0);
  }
  transW(encw[2], 128, 4, 128);
  {
    dim3 g((O + 255) / 256, 1);
    gemm_wmma<0, 2><<<g, 256, 0, stream>>>(
        pooledb, (const bf16_t*)nullptr, (const bf16_t*)nullptr, s_idx, o_idx, 0,
        Btw, encb[2], O, 4, 128,
        (bf16_t*)nullptr, 0, (float*)nullptr, (bf16_t*)nullptr, (float*)d_out, 0, 0);
  }
}
